// PointNet2Encoder_81819126989015
// MI455X (gfx1250) — compile-verified, matrix operand order checked by reading, not run on hardware
//
#include <hip/hip_runtime.h>
#include <hip/hip_bf16.h>

typedef __attribute__((ext_vector_type(16))) _Float16 v16h;
typedef __attribute__((ext_vector_type(8)))  _Float16 h8;
typedef __attribute__((ext_vector_type(8)))  float    v8f;

#define BIGF 1e10f

__device__ __forceinline__ v16h join16(h8 lo, h8 hi) {
    return __builtin_shufflevector(lo, hi, 0, 1, 2, 3, 4, 5, 6, 7,
                                           8, 9, 10, 11, 12, 13, 14, 15);
}

// ---------------------------------------------------------------------------
// WMMA GEMM: C[M,N] = relu(A[M,Kp]h16 @ Wt[N,Kp]h16^T + bias[N])
// A rows padded to Kp (multiple of 32 halves, 64B) with zeros; Wt is the
// weight pre-transposed [N][Kp] f16, zero-padded. Everything is aligned so
// each fragment is two global_load_b128 per lane; no LDS, no guards, EXEC
// all-ones at every WMMA. grid=(M/32, N/64), block=128 (4 waves); each wave
// owns a 32(M)x16(N) strip -> 2 accumulators, 2 v_wmma per K-step.
// ---------------------------------------------------------------------------
template <typename OutT>
__global__ void gemm_wmma(const _Float16* __restrict__ A, int lda,
                          const _Float16* __restrict__ Wt, int ldw,
                          const float* __restrict__ bias,
                          OutT* __restrict__ C, int ldc,
                          int M, int N, int Kp)
{
    const int tid  = threadIdx.x;
    const int wave = tid >> 5;
    const int lane = tid & 31;
    const int hl   = lane >> 4;      // lane-group: ISA 16-bit operand layout
    const int l    = lane & 15;
    const int m0   = blockIdx.x * 32;
    const int n0   = blockIdx.y * 64 + wave * 16;

    const _Float16* arow0 = A  + (size_t)(m0 + l)      * lda;
    const _Float16* arow1 = A  + (size_t)(m0 + 16 + l) * lda;
    const _Float16* brow  = Wt + (size_t)(n0 + l)      * ldw;
    const int koff = hl * 8;   // lanes 0-15: K {0..7,16..23}; 16-31: {8..15,24..31}

    v8f acc0 = {}, acc1 = {};

    for (int k0 = 0; k0 < Kp; k0 += 32) {
        __builtin_prefetch((const void*)(arow0 + k0 + 128), 0, 1);
        __builtin_prefetch((const void*)(brow  + k0 + 128), 0, 1);

        v16h bf = join16(*(const h8*)(brow + k0 + koff),
                         *(const h8*)(brow + k0 + 16 + koff));
        v16h a0 = join16(*(const h8*)(arow0 + k0 + koff),
                         *(const h8*)(arow0 + k0 + 16 + koff));
        v16h a1 = join16(*(const h8*)(arow1 + k0 + koff),
                         *(const h8*)(arow1 + k0 + 16 + koff));

        acc0 = __builtin_amdgcn_wmma_f32_16x16x32_f16(false, a0, false, bf,
                                                      (short)0, acc0, false, false);
        acc1 = __builtin_amdgcn_wmma_f32_16x16x32_f16(false, a1, false, bf,
                                                      (short)0, acc1, false, false);
    }

    // C/D f32 layout: N = lane%16, M = r + 8*(lane/16)
    const float bcol  = bias[n0 + l];
    const int   rbase = hl * 8;
#pragma unroll
    for (int r = 0; r < 8; ++r) {
        float v0 = fmaxf(acc0[r] + bcol, 0.0f);
        float v1 = fmaxf(acc1[r] + bcol, 0.0f);
        C[(size_t)(m0 + rbase + r)      * ldc + n0 + l] = (OutT)v0;
        C[(size_t)(m0 + 16 + rbase + r) * ldc + n0 + l] = (OutT)v1;
    }
}

// Weight prep: Wt[n*Kp + k] = (f16) W[k*N + n], zero pad k>=K
__global__ void wt_prep_kernel(const float* __restrict__ W, _Float16* __restrict__ Wt,
                               int K, int N, int Kp)
{
    int t = blockIdx.x * blockDim.x + threadIdx.x;
    if (t >= N * Kp) return;
    int n = t / Kp, k = t % Kp;
    Wt[t] = (k < K) ? (_Float16)W[(size_t)k * N + n] : (_Float16)0.0f;
}

// ---------------------------------------------------------------------------
// Farthest point sampling: one block, serial over M-1 iterations (LDS dists).
// ---------------------------------------------------------------------------
__global__ void fps_kernel(const float* __restrict__ pos, int* __restrict__ sel,
                           int N, int M)
{
    __shared__ float dists[4096];
    __shared__ float rv[256];
    __shared__ int   ri[256];
    __shared__ int   s_last;
    const int tid = threadIdx.x;

    for (int j = tid; j < N; j += 256) dists[j] = BIGF;
    if (tid == 0) { sel[0] = 0; s_last = 0; }
    __syncthreads();

    for (int it = 1; it < M; ++it) {
        int last = s_last;
        float lx = pos[last * 3 + 0], ly = pos[last * 3 + 1], lz = pos[last * 3 + 2];
        float bv = -1.0f; int bi = 0;
        for (int j = tid; j < N; j += 256) {
            float dx = pos[j * 3 + 0] - lx;
            float dy = pos[j * 3 + 1] - ly;
            float dz = pos[j * 3 + 2] - lz;
            float d  = dx * dx + dy * dy + dz * dz;
            float mn = fminf(dists[j], d);
            dists[j] = mn;
            if (mn > bv) { bv = mn; bi = j; }
        }
        rv[tid] = bv; ri[tid] = bi;
        __syncthreads();
        for (int s = 128; s > 0; s >>= 1) {
            if (tid < s) {
                if (rv[tid + s] > rv[tid] ||
                    (rv[tid + s] == rv[tid] && ri[tid + s] < ri[tid])) {
                    rv[tid] = rv[tid + s]; ri[tid] = ri[tid + s];
                }
            }
            __syncthreads();
        }
        if (tid == 0) { s_last = ri[0]; sel[it] = ri[0]; }
        __syncthreads();
    }
}

__global__ void gather_pos_kernel(const float* __restrict__ pos,
                                  const int* __restrict__ sel,
                                  float* __restrict__ out, int M)
{
    int t = blockIdx.x * blockDim.x + threadIdx.x;
    if (t >= M * 3) return;
    out[t] = pos[sel[t / 3] * 3 + (t % 3)];
}

// ---------------------------------------------------------------------------
// Ball query: per center keep 32 smallest masked distances.
// ---------------------------------------------------------------------------
__global__ void ball_query_kernel(const float* __restrict__ centers,
                                  const float* __restrict__ pos,
                                  int* __restrict__ out,
                                  int M, int N, float r2)
{
    int m = blockIdx.x * blockDim.x + threadIdx.x;
    if (m >= M) return;
    const int K = 32;
    float bd[32]; int bi[32];
    for (int k = 0; k < K; ++k) { bd[k] = BIGF; bi[k] = 0; }
    float cx = centers[m * 3 + 0], cy = centers[m * 3 + 1], cz = centers[m * 3 + 2];
    for (int j = 0; j < N; ++j) {
        float dx = pos[j * 3 + 0] - cx;
        float dy = pos[j * 3 + 1] - cy;
        float dz = pos[j * 3 + 2] - cz;
        float d  = dx * dx + dy * dy + dz * dz;
        float dm = (d <= r2) ? d : BIGF;
        if (dm < bd[K - 1]) {
            int k = K - 1;
            while (k > 0 && bd[k - 1] > dm) { bd[k] = bd[k - 1]; bi[k] = bi[k - 1]; --k; }
            bd[k] = dm; bi[k] = j;
        }
    }
    int i0 = bi[0];
    for (int k = 0; k < K; ++k) out[m * K + k] = (bd[k] < BIGF * 0.5f) ? bi[k] : i0;
}

// grouped rows (f16, padded width Cp): [pos[idx]-center | feat[idx] | 0-pad]
__global__ void group_kernel(const float* __restrict__ pos,
                             const float* __restrict__ centers,
                             const int* __restrict__ idx,
                             const _Float16* __restrict__ feat, int Cf,
                             _Float16* __restrict__ out, int MK, int K, int Cp)
{
    int t = blockIdx.x * blockDim.x + threadIdx.x;
    if (t >= MK * Cp) return;
    int r = t / Cp, c = t % Cp;
    int m = r / K;
    int j = idx[r];
    float v = 0.0f;
    if (c < 3)           v = pos[j * 3 + c] - centers[m * 3 + c];
    else if (c < 3 + Cf) v = (float)feat[(size_t)j * Cf + (c - 3)];
    out[t] = (_Float16)v;
}

__global__ void maxpool_kernel(const _Float16* __restrict__ h, _Float16* __restrict__ out,
                               int M, int K, int C)
{
    int t = blockIdx.x * blockDim.x + threadIdx.x;
    if (t >= M * C) return;
    int m = t / C, c = t % C;
    float v = -BIGF;
    for (int k = 0; k < K; ++k)
        v = fmaxf(v, (float)h[((size_t)m * K + k) * C + c]);
    out[t] = (_Float16)v;
}

__global__ void colmax_kernel(const _Float16* __restrict__ h, _Float16* __restrict__ out,
                              int R, int C)
{
    int c = blockIdx.x * blockDim.x + threadIdx.x;
    if (c >= C) return;
    float v = -BIGF;
    for (int r = 0; r < R; ++r) v = fmaxf(v, (float)h[(size_t)r * C + c]);
    out[c] = (_Float16)v;
}

// out f16 [M,Cp]: c<Ca -> a (f32 or f16, optionally broadcast), then b (f16), then 0
template <typename TA>
__global__ void concat_kernel(const TA* __restrict__ a, int Ca, int bcastA,
                              const _Float16* __restrict__ b, int Cb,
                              _Float16* __restrict__ out, int M, int Cp)
{
    int t = blockIdx.x * blockDim.x + threadIdx.x;
    if (t >= M * Cp) return;
    int m = t / Cp, c = t % Cp;
    float v = 0.0f;
    if (c < Ca)           v = (float)(bcastA ? a[c] : a[(size_t)m * Ca + c]);
    else if (c < Ca + Cb) v = (float)b[(size_t)m * Cb + (c - Ca)];
    out[t] = (_Float16)v;
}

// 3-NN + inverse-distance weights
__global__ void knn3_kernel(const float* __restrict__ dst, const float* __restrict__ src,
                            int* __restrict__ idx3, float* __restrict__ w3,
                            int Nd, int Ns)
{
    int n = blockIdx.x * blockDim.x + threadIdx.x;
    if (n >= Nd) return;
    float d0 = 1e30f, d1 = 1e30f, d2 = 1e30f;
    int   i0 = 0, i1 = 0, i2 = 0;
    float x = dst[n * 3 + 0], y = dst[n * 3 + 1], z = dst[n * 3 + 2];
    for (int j = 0; j < Ns; ++j) {
        float dx = src[j * 3 + 0] - x;
        float dy = src[j * 3 + 1] - y;
        float dz = src[j * 3 + 2] - z;
        float d  = dx * dx + dy * dy + dz * dz;
        if (d < d0)      { d2 = d1; i2 = i1; d1 = d0; i1 = i0; d0 = d; i0 = j; }
        else if (d < d1) { d2 = d1; i2 = i1; d1 = d;  i1 = j; }
        else if (d < d2) { d2 = d;  i2 = j; }
    }
    float w0 = 1.0f / fmaxf(d0, 1e-10f);
    float w1 = 1.0f / fmaxf(d1, 1e-10f);
    float w2 = 1.0f / fmaxf(d2, 1e-10f);
    float s = w0 + w1 + w2;
    idx3[n * 3 + 0] = i0; idx3[n * 3 + 1] = i1; idx3[n * 3 + 2] = i2;
    w3[n * 3 + 0] = w0 / s; w3[n * 3 + 1] = w1 / s; w3[n * 3 + 2] = w2 / s;
}

// out f16 [Nd,Cp]: c<Cs -> skip, c<Cs+Cf -> idw-interp of feat, else 0
__global__ void interp_concat_kernel(const _Float16* __restrict__ skip, int Cs,
                                     const _Float16* __restrict__ feat, int Cf,
                                     const int* __restrict__ idx3,
                                     const float* __restrict__ w3,
                                     _Float16* __restrict__ out, int Nd, int Cp)
{
    int t = blockIdx.x * blockDim.x + threadIdx.x;
    if (t >= Nd * Cp) return;
    int n = t / Cp, c = t % Cp;
    float v = 0.0f;
    if (c < Cs) {
        v = (float)skip[(size_t)n * Cs + c];
    } else if (c < Cs + Cf) {
        int cc = c - Cs;
        v = w3[n * 3 + 0] * (float)feat[(size_t)idx3[n * 3 + 0] * Cf + cc]
          + w3[n * 3 + 1] * (float)feat[(size_t)idx3[n * 3 + 1] * Cf + cc]
          + w3[n * 3 + 2] * (float)feat[(size_t)idx3[n * 3 + 2] * Cf + cc];
    }
    out[t] = (_Float16)v;
}

// g[b][c] = mean over N rows of F[b,:,c]
__global__ void mean_kernel(const float* __restrict__ F, float* __restrict__ g,
                            int N, int C)
{
    int b = blockIdx.x;
    int c = threadIdx.x;
    if (c >= C) return;
    const float* p = F + (size_t)b * N * C + c;
    float s = 0.0f;
    for (int n = 0; n < N; ++n) s += p[(size_t)n * C];
    g[b * C + c] = s / (float)N;
}

// ---------------------------------------------------------------------------
// Host orchestration
// ---------------------------------------------------------------------------
extern "C" void kernel_launch(void* const* d_in, const int* in_sizes, int n_in,
                              void* d_out, int out_size, void* d_ws, size_t ws_size,
                              hipStream_t stream)
{
    (void)in_sizes; (void)n_in; (void)out_size; (void)ws_size;

    // d_in (setup_inputs dict insertion order): pts, then (W,b) of
    // sa1 x3, sa2 x3, sa3 x3, fp3 x2, fp2 x2, fp1 x3  -> 33 pointers
    const float* P[33];
    for (int i = 0; i < 33; ++i) P[i] = (const float*)d_in[i];
    const float* pts = P[0];

    const int B = 8, NPTS = 4096;
    auto KP   = [](int K) { return (K + 31) & ~31; };
    auto blks = [](long n) { return (int)((n + 255) / 256); };

    char* wp = (char*)d_ws;
    auto carve = [&](size_t bytes) -> void* {
        void* p = (void*)wp;
        wp += (bytes + 255) & ~(size_t)255;
        return p;
    };

    // layer table: {weight input idx, K, N}; bias is idx+1
    struct LW { int pi, K, N; };
    const LW lw[16] = {
        {1, 3, 64},    {3, 64, 64},   {5, 64, 128},              // sa1
        {7, 131, 128}, {9, 128, 128}, {11, 128, 256},            // sa2
        {13, 259, 256},{15, 256, 512},{17, 512, 1024},           // sa3
        {19, 1280, 256},{21, 256, 256},                          // fp3
        {23, 384, 256},{25, 256, 128},                           // fp2
        {27, 128, 128},{29, 128, 128},{31, 128, 256}             // fp1
    };
    _Float16* Wt[16];
    for (int i = 0; i < 16; ++i) {
        int Kp = KP(lw[i].K);
        Wt[i] = (_Float16*)carve((size_t)lw[i].N * Kp * sizeof(_Float16));
        wt_prep_kernel<<<blks((long)lw[i].N * Kp), 256, 0, stream>>>(
            P[lw[i].pi], Wt[i], lw[i].K, lw[i].N, Kp);
    }

    int*      sel1  = (int*)      carve(1024 * sizeof(int));
    float*    p1    = (float*)    carve(1024 * 3 * sizeof(float));
    int*      idx1  = (int*)      carve(1024 * 32 * sizeof(int));
    _Float16* g1in  = (_Float16*) carve((size_t)32768 * 32 * sizeof(_Float16));
    _Float16* bufA  = (_Float16*) carve((size_t)32768 * 128 * sizeof(_Float16));
    _Float16* bufB  = (_Float16*) carve((size_t)32768 * 128 * sizeof(_Float16));
    _Float16* f1    = (_Float16*) carve((size_t)1024 * 128 * sizeof(_Float16));
    int*      sel2  = (int*)      carve(256 * sizeof(int));
    float*    p2    = (float*)    carve(256 * 3 * sizeof(float));
    int*      idx2  = (int*)      carve(256 * 32 * sizeof(int));
    _Float16* g2in  = (_Float16*) carve((size_t)8192 * 160 * sizeof(_Float16));
    _Float16* f2    = (_Float16*) carve((size_t)256 * 256 * sizeof(_Float16));
    _Float16* sa3in = (_Float16*) carve((size_t)256 * 288 * sizeof(_Float16));
    _Float16* gvec  = (_Float16*) carve(1024 * sizeof(_Float16));
    _Float16* fp3in = (_Float16*) carve((size_t)256 * 1280 * sizeof(_Float16));
    _Float16* f2u   = (_Float16*) carve((size_t)256 * 256 * sizeof(_Float16));
    int*      knn_i = (int*)      carve((size_t)4096 * 3 * sizeof(int));
    float*    knn_w = (float*)    carve((size_t)4096 * 3 * sizeof(float));
    _Float16* fp2in = (_Float16*) carve((size_t)1024 * 384 * sizeof(_Float16));
    _Float16* f1u   = (_Float16*) carve((size_t)1024 * 128 * sizeof(_Float16));
    _Float16* fp1in = (_Float16*) carve((size_t)4096 * 128 * sizeof(_Float16));

    // GEMM to f16 activation buffer (output stride = N, already 32-aligned)
    auto gemmh = [&](const _Float16* A, int li, _Float16* C, int M) {
        int Kp = KP(lw[li].K), N = lw[li].N;
        dim3 grid(M / 32, N / 64);
        gemm_wmma<_Float16><<<grid, 128, 0, stream>>>(
            A, Kp, Wt[li], Kp, P[lw[li].pi + 1], C, N, M, N, Kp);
    };

    for (int b = 0; b < B; ++b) {
        const float* pos = pts + (size_t)b * NPTS * 3;
        float* f0_out = (float*)d_out + (size_t)b * NPTS * 256;

        // ---- SA1: FPS 4096->1024, ball r=0.2 K=32, MLP [3,64,64,128], maxpool
        fps_kernel<<<1, 256, 0, stream>>>(pos, sel1, NPTS, 1024);
        gather_pos_kernel<<<blks(1024 * 3), 256, 0, stream>>>(pos, sel1, p1, 1024);
        ball_query_kernel<<<(1024 + 63) / 64, 64, 0, stream>>>(p1, pos, idx1, 1024, NPTS, 0.04f);
        group_kernel<<<blks((long)32768 * 32), 256, 0, stream>>>(pos, p1, idx1, nullptr, 0,
                                                                 g1in, 32768, 32, 32);
        gemmh(g1in, 0, bufA, 32768);
        gemmh(bufA, 1, bufB, 32768);
        gemmh(bufB, 2, bufA, 32768);
        maxpool_kernel<<<blks(1024 * 128), 256, 0, stream>>>(bufA, f1, 1024, 32, 128);

        // ---- SA2: FPS 1024->256, ball r=0.4 K=32, MLP [131,128,128,256], maxpool
        fps_kernel<<<1, 256, 0, stream>>>(p1, sel2, 1024, 256);
        gather_pos_kernel<<<blks(256 * 3), 256, 0, stream>>>(p1, sel2, p2, 256);
        ball_query_kernel<<<(256 + 63) / 64, 64, 0, stream>>>(p2, p1, idx2, 256, 1024, 0.16f);
        group_kernel<<<blks((long)8192 * 160), 256, 0, stream>>>(p1, p2, idx2, f1, 128,
                                                                 g2in, 8192, 32, 160);
        gemmh(g2in, 3, bufA, 8192);
        gemmh(bufA, 4, bufB, 8192);
        gemmh(bufB, 5, bufA, 8192);
        maxpool_kernel<<<blks(256 * 256), 256, 0, stream>>>(bufA, f2, 256, 32, 256);

        // ---- SA3: MLP [259,256,512,1024] on concat(p2,f2), global max
        concat_kernel<float><<<blks(256 * 288), 256, 0, stream>>>(p2, 3, 0, f2, 256,
                                                                  sa3in, 256, 288);
        gemmh(sa3in, 6, bufB, 256);
        gemmh(bufB, 7, bufA, 256);
        gemmh(bufA, 8, bufB, 256);
        colmax_kernel<<<blks(1024), 256, 0, stream>>>(bufB, gvec, 256, 1024);

        // ---- FP3: concat(broadcast g, f2) -> MLP [1280,256,256]
        concat_kernel<_Float16><<<blks(256 * 1280), 256, 0, stream>>>(gvec, 1024, 1, f2, 256,
                                                                      fp3in, 256, 1280);
        gemmh(fp3in, 9, bufA, 256);
        gemmh(bufA, 10, f2u, 256);

        // ---- FP2: 3-NN p2->p1, concat(f1, interp(f2u)) -> MLP [384,256,128]
        knn3_kernel<<<(1024 + 63) / 64, 64, 0, stream>>>(p1, p2, knn_i, knn_w, 1024, 256);
        interp_concat_kernel<<<blks(1024 * 384), 256, 0, stream>>>(f1, 128, f2u, 256,
                                                                   knn_i, knn_w, fp2in, 1024, 384);
        gemmh(fp2in, 11, bufA, 1024);
        gemmh(bufA, 12, f1u, 1024);

        // ---- FP1: 3-NN p1->pos, interp(f1u) -> MLP [128,128,128,256] -> f32 out
        knn3_kernel<<<(4096 + 63) / 64, 64, 0, stream>>>(pos, p1, knn_i, knn_w, 4096, 1024);
        interp_concat_kernel<<<blks((long)4096 * 128), 256, 0, stream>>>(nullptr, 0, f1u, 128,
                                                                         knn_i, knn_w, fp1in, 4096, 128);
        gemmh(fp1in, 13, bufA, 4096);
        gemmh(bufA, 14, bufB, 4096);
        {   // final layer writes f32 straight into d_out
            int li = 15, Kp = KP(lw[li].K), N = lw[li].N;
            dim3 grid(4096 / 32, N / 64);
            gemm_wmma<float><<<grid, 128, 0, stream>>>(
                bufB, Kp, Wt[li], Kp, P[lw[li].pi + 1], f0_out, N, 4096, N, Kp);
        }
    }

    // global mean over points -> g [8,256], appended after F_geo in d_out
    mean_kernel<<<B, 256, 0, stream>>>((const float*)d_out,
                                       (float*)d_out + (size_t)B * NPTS * 256,
                                       NPTS, 256);
}
